// MultiHeadAttention_28140625724248
// MI455X (gfx1250) — compile-verified, matrix operand order checked by reading, not run on hardware
//
#include <hip/hip_runtime.h>
#include <math.h>
#include <stdint.h>

typedef __bf16 bf16_t;
typedef __attribute__((ext_vector_type(16))) __bf16 v16bf;
typedef __attribute__((ext_vector_type(8)))  __bf16 v8bf;
typedef __attribute__((ext_vector_type(4)))  __bf16 v4bf;
typedef __attribute__((ext_vector_type(8)))  float  v8f;
typedef __attribute__((ext_vector_type(4)))  float  v4f;
typedef __attribute__((ext_vector_type(4)))  unsigned int v4u;
typedef __attribute__((ext_vector_type(8)))  int v8i;
typedef __attribute__((ext_vector_type(4)))  int v4i;

#define BB 2
#define SS 2048
#define DD 1024
#define HH 16
#define HDIM 64
#define D3 3072
#define MROWS (BB * SS)   // 4096

// ---------------------------------------------------------------------------
// Tensor Data Mover availability (arity differs across toolchains)
// ---------------------------------------------------------------------------
#if __has_builtin(__builtin_amdgcn_tensor_load_to_lds) && \
    __has_builtin(__builtin_amdgcn_s_wait_tensorcnt)
#  define HAVE_TDM 1
#  if __has_include(<hip/amd_detail/amd_gfx1250_TDM.h>)
#    define TDM_6ARG 1
#  endif
#else
#  define HAVE_TDM 0
#endif

#if HAVE_TDM
// Issue a TDM 2D tile load (bf16 elements) into LDS.  D# layout per CDNA5 ISA
// section 8.3-8.6: group0 = {count/flags, lds_addr, global_addr, type=2},
// group1 = {data_size=2B, tensor_dim0/1, tile_dim0/1, dim0_stride}.
__device__ __forceinline__ void tdm_load_tile_2d(unsigned lds_off,
                                                 const void* gptr,
                                                 unsigned tensor_d0,
                                                 unsigned tensor_d1,
                                                 unsigned tile_d0,
                                                 unsigned tile_d1,
                                                 unsigned stride0) {
  unsigned long long ga = (unsigned long long)(uintptr_t)gptr;
  v4u g0;
  g0.x = 1u;                                           // count=1 (valid)
  g0.y = lds_off;                                      // lds_addr
  g0.z = (unsigned)(ga & 0xffffffffu);                 // global_addr[31:0]
  g0.w = (unsigned)((ga >> 32) & 0x1ffffffu) | (2u << 30);  // addr[56:32]|type=2
  v8i g1;
  g1[0] = (int)(1u << 16);                             // data_size=1 -> 2 bytes
  g1[1] = (int)((tensor_d0 & 0xffffu) << 16);          // tensor_dim0[15:0]
  g1[2] = (int)((tensor_d0 >> 16) | ((tensor_d1 & 0xffffu) << 16));
  g1[3] = (int)((tensor_d1 >> 16) | (tile_d0 << 16));  // tile_dim0
  g1[4] = (int)(tile_d1 & 0xffffu);                    // tile_dim1 (tile_dim2=0)
  g1[5] = (int)stride0;                                // tensor_dim0_stride lo32
  g1[6] = 0;
  g1[7] = 0;
  v4i z4 = {0, 0, 0, 0};
#if defined(TDM_6ARG)
  v8i z8 = {0, 0, 0, 0, 0, 0, 0, 0};
  __builtin_amdgcn_tensor_load_to_lds(g0, g1, z4, z4, z8, 0);
#else
  __builtin_amdgcn_tensor_load_to_lds(g0, g1, z4, z4, 0);
#endif
}
#endif  // HAVE_TDM

// ---------------------------------------------------------------------------
// Fragment loaders: 16-bit A/B fragment (16 rows x 32 K) per CDNA5 ISA layout.
// lane L: row = L&15, kk = (L>>4)*8 ; VGPR0-3 hold K=kk..kk+7, VGPR4-7 hold
// K=kk+16..kk+23.
// ---------------------------------------------------------------------------
__device__ __forceinline__ v16bf load_frag_p(const bf16_t* p) {
  union { v16bf v; v8bf h[2]; } u;
  u.h[0] = *(const v8bf*)p;         // K = +0  .. +7   (16B load)
  u.h[1] = *(const v8bf*)(p + 16);  // K = +16 .. +23  (16B load)
  return u.v;
}

__device__ __forceinline__ v16bf load_frag(const bf16_t* base, int ld,
                                           int row16, int kBase) {
  int lane = threadIdx.x & 31;
  int r  = lane & 15;
  int kk = (lane >> 4) << 3;
  return load_frag_p(base + (size_t)(row16 + r) * (size_t)ld + kBase + kk);
}

__device__ __forceinline__ v8f wmma_bf16(v16bf a, v16bf b, v8f c) {
  return __builtin_amdgcn_wmma_f32_16x16x32_bf16(false, a, false, b,
                                                 (short)0, c, false, false);
}

// ---------------------------------------------------------------------------
// Pack kernels
// ---------------------------------------------------------------------------
__global__ void mha_cvt_bf16(const float* __restrict__ src,
                             bf16_t* __restrict__ dst, int n4) {
  int i = blockIdx.x * blockDim.x + threadIdx.x;
  if (i >= n4) return;
  v4f a = ((const v4f*)src)[i];
  v4bf o;
  o[0] = (bf16_t)a[0]; o[1] = (bf16_t)a[1];
  o[2] = (bf16_t)a[2]; o[3] = (bf16_t)a[3];
  ((v4bf*)dst)[i] = o;
}

// dst[c*rows + r] = (bf16) src[r*cols + c]   (LDS-tiled, conflict-free)
__global__ void mha_transpose_bf16(const float* __restrict__ src,
                                   bf16_t* __restrict__ dst,
                                   int rows, int cols) {
  __shared__ float tile[32][33];
  int c0 = blockIdx.x * 32, r0 = blockIdx.y * 32;
  int tx = threadIdx.x, ty = threadIdx.y;       // 32 x 8
  #pragma unroll
  for (int j = 0; j < 32; j += 8)
    tile[ty + j][tx] = src[(size_t)(r0 + ty + j) * cols + c0 + tx];
  __syncthreads();
  #pragma unroll
  for (int j = 0; j < 32; j += 8)
    dst[(size_t)(c0 + ty + j) * rows + r0 + tx] = (bf16_t)tile[tx][ty + j];
}

// ---------------------------------------------------------------------------
// Software-pipelined 32x32-per-wave GEMM core.  Only TWO per-lane base
// pointers (A, B); the second row-tile of each sits at a constant +16*DD
// element offset that folds into the global-load 24-bit immediate, and the
// k-offset is an induction variable -> minimal address registers, no spills.
// Next iteration's 8 b128 loads are issued before the current 4 WMMAs.
// ---------------------------------------------------------------------------
#define RSTEP ((size_t)16 * DD)   // +16 rows, in elements (32KB, fits imm24)

#define GEMM_CORE(Abase, Bbase, mBase, nBase)                                 \
  int lan_ = threadIdx.x & 31;                                                \
  int fr_  = lan_ & 15;                                                       \
  int fk_  = (lan_ >> 4) << 3;                                                \
  const bf16_t* pa_ = (Abase) + (size_t)((mBase) + fr_) * DD + fk_;           \
  const bf16_t* pb_ = (Bbase) + (size_t)((nBase) + fr_) * DD + fk_;           \
  v8f acc00 = {}, acc01 = {}, acc10 = {}, acc11 = {};                         \
  v16bf a0 = load_frag_p(pa_);                                                \
  v16bf a1 = load_frag_p(pa_ + RSTEP);                                        \
  v16bf b0 = load_frag_p(pb_);                                                \
  v16bf b1 = load_frag_p(pb_ + RSTEP);                                        \
  for (int k = 32; k < DD; k += 32) {                                         \
    v16bf an0 = load_frag_p(pa_ + k);                                         \
    v16bf an1 = load_frag_p(pa_ + RSTEP + k);                                 \
    v16bf bn0 = load_frag_p(pb_ + k);                                         \
    v16bf bn1 = load_frag_p(pb_ + RSTEP + k);                                 \
    acc00 = wmma_bf16(a0, b0, acc00);                                         \
    acc01 = wmma_bf16(a0, b1, acc01);                                         \
    acc10 = wmma_bf16(a1, b0, acc10);                                         \
    acc11 = wmma_bf16(a1, b1, acc11);                                         \
    a0 = an0; a1 = an1; b0 = bn0; b1 = bn1;                                   \
  }                                                                           \
  acc00 = wmma_bf16(a0, b0, acc00);                                           \
  acc01 = wmma_bf16(a0, b1, acc01);                                           \
  acc10 = wmma_bf16(a1, b0, acc10);                                           \
  acc11 = wmma_bf16(a1, b1, acc11);

// ---------------------------------------------------------------------------
// QKV projection:  qkv = x @ W_qkv + b_qkv ; scatter to Q,K (B,H,S,64) and
// V^T (B,H,64,S), all bf16.  One wave = 32x32 output tile, 8 waves per block.
// ---------------------------------------------------------------------------
__global__ void __launch_bounds__(256, 1)
mha_qkv_gemm(const bf16_t* __restrict__ xb,      // (M, D)
             const bf16_t* __restrict__ wqkv_t,  // (3D, D)
             const float*  __restrict__ bqkv,    // (3D)
             bf16_t* __restrict__ Qb,
             bf16_t* __restrict__ Kb,
             bf16_t* __restrict__ Vt) {
  int mBase = blockIdx.x * 32;
  int nBase = (blockIdx.y * 8 + (threadIdx.x >> 5)) * 32;

  GEMM_CORE(xb, wqkv_t, mBase, nBase)

  int lane = threadIdx.x & 31;
  int half = lane >> 4;
  int nlo  = lane & 15;
  v8f* accs[2][2] = {{&acc00, &acc01}, {&acc10, &acc11}};
  #pragma unroll
  for (int mt = 0; mt < 2; ++mt) {
    #pragma unroll
    for (int nt = 0; nt < 2; ++nt) {
      int n  = nBase + nt * 16 + nlo;
      int hh = n / (3 * HDIM);
      int r  = n % (3 * HDIM);
      float bias = bqkv[n];
      #pragma unroll
      for (int i = 0; i < 8; ++i) {
        int row = mBase + mt * 16 + i + 8 * half;   // global row over B*S
        int b_  = row / SS, s = row % SS;
        bf16_t v = (bf16_t)((*accs[mt][nt])[i] + bias);
        size_t head = (size_t)(b_ * HH + hh);
        if (r < HDIM)          Qb[(head * SS + s) * HDIM + r]              = v;
        else if (r < 2 * HDIM) Kb[(head * SS + s) * HDIM + (r - HDIM)]     = v;
        else                   Vt[(head * HDIM + (r - 2 * HDIM)) * SS + s] = v;
      }
    }
  }
}

// ---------------------------------------------------------------------------
// Flash attention: block = 4 waves (4 query tiles of one (b,h)); the K tile
// (32x64) and V^T tile (64x32) are staged in LDS once per block -- via the
// Tensor Data Mover when available (tensor_load_to_lds + s_wait_tensorcnt),
// else a cooperative vectorized copy.  Online softmax in registers; P goes
// through a per-wave LDS round trip to become an A fragment. 8 wmma/key tile.
// ---------------------------------------------------------------------------
__global__ void mha_flash_attn(const bf16_t* __restrict__ Qb,
                               const bf16_t* __restrict__ Kb,
                               const bf16_t* __restrict__ Vt,
                               const float*  __restrict__ mask, // (S,S)
                               bf16_t* __restrict__ ctxb) {     // (M, D)
  __shared__ __align__(16) bf16_t Klds[32 * HDIM];   // key-major, ld=64
  __shared__ __align__(16) bf16_t Vlds[HDIM * 32];   // hd-major,  ld=32
  __shared__ __align__(16) bf16_t Plds[4][16 * 32];

  int bh    = blockIdx.x;                       // b*H + h
  int wave  = threadIdx.x >> 5;
  int qTile = blockIdx.y * 4 + wave;
  int lane  = threadIdx.x & 31;
  int half  = lane >> 4;
  int nlo   = lane & 15;
  int q0    = qTile * 16;

  const bf16_t* Qh = Qb + (size_t)bh * SS * HDIM;
  const bf16_t* Kh = Kb + (size_t)bh * SS * HDIM;
  const bf16_t* Vh = Vt + (size_t)bh * HDIM * SS;

  // Q fragments for both K-steps over head dim (resident all loop long)
  v16bf qf0 = load_frag(Qh, HDIM, q0, 0);
  v16bf qf1 = load_frag(Qh, HDIM, q0, 32);

  v8f acc0 = {}, acc1 = {}, acc2 = {}, acc3 = {};
  float rowmax[8], rowsum[8];
  #pragma unroll
  for (int i = 0; i < 8; ++i) { rowmax[i] = -3.0e38f; rowsum[i] = 0.0f; }

  const float scale = 0.125f;   // 1/sqrt(64)

  for (int kb = 0; kb < SS; kb += 32) {
    __syncthreads();            // previous tile fully consumed
    // ---- stage K (32x64) and V^T (64x32) tiles into LDS ----
#if HAVE_TDM
    if (wave == 0) {
      tdm_load_tile_2d((unsigned)(uintptr_t)(void*)Klds,
                       Kh + (size_t)kb * HDIM,
                       /*tensor_d0=*/HDIM, /*tensor_d1=*/SS,
                       /*tile_d0=*/HDIM, /*tile_d1=*/32, /*stride0=*/HDIM);
      tdm_load_tile_2d((unsigned)(uintptr_t)(void*)Vlds,
                       Vh + kb,
                       /*tensor_d0=*/SS, /*tensor_d1=*/HDIM,
                       /*tile_d0=*/32, /*tile_d1=*/HDIM, /*stride0=*/SS);
      __builtin_amdgcn_s_wait_tensorcnt(0);
    }
#else
    {
      const v8bf* ksrc = (const v8bf*)(Kh + (size_t)kb * HDIM);
      v8bf* kdst = (v8bf*)Klds;
      v8bf* vdst = (v8bf*)Vlds;
      #pragma unroll
      for (int u = threadIdx.x; u < 256; u += 128) {
        kdst[u] = ksrc[u];                          // K tile is contiguous
        int r = u >> 2, c = u & 3;                  // V^T rows strided by S
        vdst[u] = *(const v8bf*)(Vh + (size_t)r * SS + kb + c * 8);
      }
    }
#endif
    __syncthreads();

    // ---- scores: 16 queries x 32 keys (two 16x16 C tiles, 4 wmma) ----
    v8f s0 = {}, s1 = {};
    s0 = wmma_bf16(qf0, load_frag(Klds, HDIM, 0, 0),   s0);
    s0 = wmma_bf16(qf1, load_frag(Klds, HDIM, 0, 32),  s0);
    s1 = wmma_bf16(qf0, load_frag(Klds, HDIM, 16, 0),  s1);
    s1 = wmma_bf16(qf1, load_frag(Klds, HDIM, 16, 32), s1);

    // ---- online softmax (row stats live per VGPR-slot i, per half) ----
    #pragma unroll
    for (int i = 0; i < 8; ++i) {
      int srow = q0 + i + 8 * half;
      size_t mrow = (size_t)srow * SS + kb;
      float e0 = s0[i] * scale + mask[mrow + nlo];
      float e1 = s1[i] * scale + mask[mrow + 16 + nlo];
      float mx = fmaxf(e0, e1);
      #pragma unroll
      for (int d = 1; d <= 8; d <<= 1)
        mx = fmaxf(mx, __shfl_xor(mx, d, 32));     // reduce 16-lane half
      float mnew = fmaxf(rowmax[i], mx);
      float rsc  = __expf(rowmax[i] - mnew);
      rowmax[i]  = mnew;
      float p0 = __expf(e0 - mnew);
      float p1 = __expf(e1 - mnew);
      float ps = p0 + p1;
      #pragma unroll
      for (int d = 1; d <= 8; d <<= 1)
        ps += __shfl_xor(ps, d, 32);
      rowsum[i] = rowsum[i] * rsc + ps;
      acc0[i] *= rsc; acc1[i] *= rsc; acc2[i] *= rsc; acc3[i] *= rsc;
      int pr = (i + 8 * half) * 32;                // C-layout -> row-major
      Plds[wave][pr + nlo]      = (bf16_t)p0;
      Plds[wave][pr + 16 + nlo] = (bf16_t)p1;
    }
    // same-wave LDS ops are in-order: store->load needs no barrier
    union { v16bf v; v8bf h[2]; } up;
    {
      int kk = half << 3;
      const bf16_t* pp = &Plds[wave][nlo * 32 + kk];
      up.h[0] = *(const v8bf*)pp;
      up.h[1] = *(const v8bf*)(pp + 16);
    }
    // ---- ctx += P @ V  (V^T in LDS gives contiguous B frags; 4 wmma) ----
    acc0 = wmma_bf16(up.v, load_frag(Vlds, 32, 0,  0), acc0);
    acc1 = wmma_bf16(up.v, load_frag(Vlds, 32, 16, 0), acc1);
    acc2 = wmma_bf16(up.v, load_frag(Vlds, 32, 32, 0), acc2);
    acc3 = wmma_bf16(up.v, load_frag(Vlds, 32, 48, 0), acc3);
  }

  // ---- epilogue: ctx / rowsum -> bf16 (B,S,H*64) for out-projection ----
  int b_ = bh / HH, h_ = bh % HH;
  #pragma unroll
  for (int i = 0; i < 8; ++i) {
    int srow = q0 + i + 8 * half;
    float rinv = 1.0f / rowsum[i];
    size_t o = ((size_t)(b_ * SS + srow)) * DD + h_ * HDIM + nlo;
    ctxb[o +  0] = (bf16_t)(acc0[i] * rinv);
    ctxb[o + 16] = (bf16_t)(acc1[i] * rinv);
    ctxb[o + 32] = (bf16_t)(acc2[i] * rinv);
    ctxb[o + 48] = (bf16_t)(acc3[i] * rinv);
  }
}

// ---------------------------------------------------------------------------
// Output projection: out = ctx @ W_out + b_out (f32 result).
// Same software-pipelined 32x32-per-wave core as the QKV GEMM.
// ---------------------------------------------------------------------------
__global__ void __launch_bounds__(256, 1)
mha_out_gemm(const bf16_t* __restrict__ ctxb,   // (M, D)
             const bf16_t* __restrict__ wout_t, // (D, D)
             const float*  __restrict__ bout,   // (D)
             float* __restrict__ out) {         // (M, D)
  int mBase = blockIdx.x * 32;
  int nBase = (blockIdx.y * 8 + (threadIdx.x >> 5)) * 32;

  GEMM_CORE(ctxb, wout_t, mBase, nBase)

  int lane = threadIdx.x & 31;
  int half = lane >> 4;
  int nlo  = lane & 15;
  v8f* accs[2][2] = {{&acc00, &acc01}, {&acc10, &acc11}};
  #pragma unroll
  for (int mt = 0; mt < 2; ++mt) {
    #pragma unroll
    for (int nt = 0; nt < 2; ++nt) {
      int n = nBase + nt * 16 + nlo;
      float bias = bout[n];
      #pragma unroll
      for (int i = 0; i < 8; ++i) {
        int row = mBase + mt * 16 + i + 8 * half;
        out[(size_t)row * DD + n] = (*accs[mt][nt])[i] + bias;
      }
    }
  }
}

// ---------------------------------------------------------------------------
extern "C" void kernel_launch(void* const* d_in, const int* in_sizes, int n_in,
                              void* d_out, int out_size, void* d_ws,
                              size_t ws_size, hipStream_t stream) {
  (void)in_sizes; (void)n_in; (void)out_size; (void)ws_size;
  const float* x    = (const float*)d_in[0];
  const float* mask = (const float*)d_in[1];
  const float* Wqkv = (const float*)d_in[2];
  const float* bqkv = (const float*)d_in[3];
  const float* Wout = (const float*)d_in[4];
  const float* bout = (const float*)d_in[5];
  float* out = (float*)d_out;

  char* ws = (char*)d_ws;
  const size_t MB = 1u << 20;
  bf16_t* xb     = (bf16_t*)(ws);                 // 8 MB  (M x D)
  bf16_t* wqkv_t = (bf16_t*)(ws +  8 * MB);       // 6 MB  (3D x D)
  bf16_t* wout_t = (bf16_t*)(ws + 14 * MB);       // 2 MB  (D x D)
  bf16_t* Qb     = (bf16_t*)(ws + 16 * MB);       // 8 MB  (B,H,S,64)
  bf16_t* Kb     = (bf16_t*)(ws + 24 * MB);       // 8 MB  (B,H,S,64)
  bf16_t* Vt     = (bf16_t*)(ws + 32 * MB);       // 8 MB  (B,H,64,S)
  bf16_t* ctxb   = (bf16_t*)(ws + 40 * MB);       // 8 MB  (M x D)

  // 1) pack / transpose to bf16
  {
    int n4 = (MROWS * DD) / 4;
    mha_cvt_bf16<<<(n4 + 255) / 256, 256, 0, stream>>>(x, xb, n4);
    mha_transpose_bf16<<<dim3(D3 / 32, DD / 32), dim3(32, 8), 0, stream>>>(
        Wqkv, wqkv_t, DD, D3);
    mha_transpose_bf16<<<dim3(DD / 32, DD / 32), dim3(32, 8), 0, stream>>>(
        Wout, wout_t, DD, DD);
  }
  // 2) QKV projection: M tiles of 32; N = 3072 = gridY(12) * 8 waves * 32
  mha_qkv_gemm<<<dim3(MROWS / 32, D3 / 256), 256, 0, stream>>>(
      xb, wqkv_t, bqkv, Qb, Kb, Vt);
  // 3) fused flash attention
  mha_flash_attn<<<dim3(BB * HH, SS / 16 / 4), 128, 0, stream>>>(
      Qb, Kb, Vt, mask, ctxb);
  // 4) output projection: N = 1024 = gridY(4) * 8 waves * 32
  mha_out_gemm<<<dim3(MROWS / 32, DD / 256), 256, 0, stream>>>(
      ctxb, wout_t, bout, out);
}
